// BALayer_46119358825150
// MI455X (gfx1250) — 8-bit matrix operand order repaired and checked by reading, compiled, not run on hardware
//
#include <hip/hip_runtime.h>
#include <stdint.h>


typedef __attribute__((ext_vector_type(16))) int   v16i;
typedef __attribute__((ext_vector_type(8)))  float v8f;

#define NPTS   4096
#define NEDGE  8192
#define FP8_ONE 0x38u

#define TM 128
#define TN 64
#define LSTR 144

__global__ void zero_kernel(uint4* p, int n4) {
    int i = blockIdx.x * blockDim.x + threadIdx.x;
    if (i < n4) p[i] = make_uint4(0u, 0u, 0u, 0u);
}

__global__ void init_adj_kernel(unsigned char* A, const int* __restrict__ tracks) {
    int i = blockIdx.x * blockDim.x + threadIdx.x;
    if (i < NPTS) A[(size_t)i * NPTS + i] = (unsigned char)FP8_ONE;
    if (i < NEDGE) {
        int a = tracks[i];
        int b = tracks[NEDGE + i];
        A[(size_t)a * NPTS + b] = (unsigned char)FP8_ONE;
        A[(size_t)b * NPTS + a] = (unsigned char)FP8_ONE;
    }
}

__device__ __forceinline__ void async_copy_b128(unsigned lds_off,
                                                const unsigned char* src) {
    asm volatile("global_load_async_to_lds_b128 %0, %1, off"
                 :: "v"(lds_off), "v"(src) : "memory");
}

__global__ __launch_bounds__(256) void bool_square_kernel(
        const unsigned char* __restrict__ S, unsigned char* __restrict__ D) {
    __shared__ unsigned char ldsA[2][TM * LSTR];
    __shared__ unsigned char ldsB[2][TN * LSTR];

    const int tid  = threadIdx.x;
    const int wave = tid >> 5;
    const int lane = tid & 31;
    const int lo16 = lane & 15;
    const int hi   = lane >> 4;
    const int wm   = wave & 3;
    const int wn   = wave >> 2;
    const int rowBase = blockIdx.x * TM;
    const int colBase = blockIdx.y * TN;

    const unsigned aBase0 = (unsigned)(size_t)&ldsA[0][0];
    const unsigned aBase1 = (unsigned)(size_t)&ldsA[1][0];
    const unsigned bBase0 = (unsigned)(size_t)&ldsB[0][0];
    const unsigned bBase1 = (unsigned)(size_t)&ldsB[1][0];

    const int rA[4] = { (tid + 0)   >> 3, (tid + 256) >> 3,
                        (tid + 512) >> 3, (tid + 768) >> 3 };
    const int cA[4] = { (tid + 0)   & 7,  (tid + 256) & 7,
                        (tid + 512) & 7,  (tid + 768) & 7 };
    const int rB[2] = { (tid + 0) >> 3, (tid + 256) >> 3 };
    const int cB[2] = { (tid + 0) & 7,  (tid + 256) & 7 };

    auto stage = [&](int sel, int kbase) {
        const unsigned aB = sel ? aBase1 : aBase0;
        const unsigned bB = sel ? bBase1 : bBase0;
#pragma unroll
        for (int r = 0; r < 4; ++r)
            async_copy_b128(aB + rA[r] * LSTR + cA[r] * 16,
                            S + (size_t)(rowBase + rA[r]) * NPTS + kbase + cA[r] * 16);
#pragma unroll
        for (int r = 0; r < 2; ++r)
            async_copy_b128(bB + rB[r] * LSTR + cB[r] * 16,
                            S + (size_t)(colBase + rB[r]) * NPTS + kbase + cB[r] * 16);
    };

    const v8f zero8 = {0.f, 0.f, 0.f, 0.f, 0.f, 0.f, 0.f, 0.f};
    v8f acc[2][2];
    acc[0][0] = zero8; acc[0][1] = zero8; acc[1][0] = zero8; acc[1][1] = zero8;

    stage(0, 0);

    for (int ks = 0; ks < NPTS / 128; ++ks) {
        const int sel = ks & 1;
        if (ks + 1 < NPTS / 128) {
            stage(sel ^ 1, (ks + 1) * 128);
            asm volatile("s_wait_asynccnt 0x6" ::: "memory");
        } else {
            asm volatile("s_wait_asynccnt 0x0" ::: "memory");
        }
        __syncthreads();

        const unsigned char* curA = ldsA[sel];
        const unsigned char* curB = ldsB[sel];

        v16i afrag[2], bfrag[2];
#pragma unroll
        for (int mi = 0; mi < 2; ++mi) {
            const unsigned char* base =
                curA + (wm * 32 + mi * 16 + lo16) * LSTR + hi * 8;
#pragma unroll
            for (int j = 0; j < 8; ++j) {
                unsigned long long q = *(const unsigned long long*)(base + j * 16);
                afrag[mi][2 * j]     = (int)(unsigned)(q & 0xffffffffull);
                afrag[mi][2 * j + 1] = (int)(unsigned)(q >> 32);
            }
        }
#pragma unroll
        for (int ni = 0; ni < 2; ++ni) {
            const unsigned char* base =
                curB + (wn * 32 + ni * 16 + lo16) * LSTR + hi * 8;
#pragma unroll
            for (int j = 0; j < 8; ++j) {
                unsigned long long q = *(const unsigned long long*)(base + j * 16);
                bfrag[ni][2 * j]     = (int)(unsigned)(q & 0xffffffffull);
                bfrag[ni][2 * j + 1] = (int)(unsigned)(q >> 32);
            }
        }

#pragma unroll
        for (int mi = 0; mi < 2; ++mi)
#pragma unroll
            for (int ni = 0; ni < 2; ++ni)
                acc[mi][ni] = __builtin_amdgcn_wmma_f32_16x16x128_fp8_fp8(
                    afrag[mi], bfrag[ni], (short)0, acc[mi][ni], false, false);

        __syncthreads();
    }

#pragma unroll
    for (int mi = 0; mi < 2; ++mi)
#pragma unroll
        for (int ni = 0; ni < 2; ++ni)
#pragma unroll
            for (int v = 0; v < 8; ++v) {
                int row = rowBase + wm * 32 + mi * 16 + v + 8 * hi;
                int col = colBase + wn * 32 + ni * 16 + lo16;
                D[(size_t)row * NPTS + col] =
                    (acc[mi][ni][v] > 0.5f) ? (unsigned char)FP8_ONE
                                            : (unsigned char)0;
            }
}

__global__ void leading_kernel(const unsigned char* __restrict__ T,
                               int* __restrict__ leading) {
    int j = blockIdx.x * blockDim.x + threadIdx.x;
    if (j >= NPTS) return;
    const unsigned char* row = T + (size_t)j * NPTS;
    int res = j;
    for (int i = 0; i <= j; i += 4) {
        unsigned w = *(const unsigned*)(row + i);
        if (w) { res = i + (__builtin_ctz(w) >> 3); break; }
    }
    leading[j] = res;
}

__global__ void assoc_kernel(const int* __restrict__ leading,
                             int* __restrict__ pid, float* __restrict__ out) {
    if (threadIdx.x == 0 && blockIdx.x == 0) {
        int cnt = 0;
        for (int j = 0; j < NPTS; ++j) {
            int l = leading[j];
            if (l == j) cnt++;
            pid[j] = cnt - 1;
            out[j] = (float)pid[l];
        }
    }
}

extern "C" void kernel_launch(void* const* d_in, const int* in_sizes, int n_in,
                              void* d_out, int out_size, void* d_ws, size_t ws_size,
                              hipStream_t stream) {
    const int* tracks = (const int*)d_in[4];

    unsigned char* buf0 = (unsigned char*)d_ws;
    unsigned char* buf1 = buf0 + (size_t)NPTS * NPTS;
    int* leading = (int*)(buf1 + (size_t)NPTS * NPTS);
    int* pid     = leading + NPTS;
    float* out   = (float*)d_out;

    const int n4 = (int)((size_t)NPTS * NPTS / 16);
    zero_kernel<<<(n4 + 255) / 256, 256, 0, stream>>>((uint4*)buf0, n4);
    init_adj_kernel<<<(NEDGE + 255) / 256, 256, 0, stream>>>(buf0, tracks);

    dim3 grid(NPTS / TM, NPTS / TN);
    bool_square_kernel<<<grid, 256, 0, stream>>>(buf0, buf1);
    bool_square_kernel<<<grid, 256, 0, stream>>>(buf1, buf0);
    bool_square_kernel<<<grid, 256, 0, stream>>>(buf0, buf1);
    bool_square_kernel<<<grid, 256, 0, stream>>>(buf1, buf0);

    leading_kernel<<<(NPTS + 255) / 256, 256, 0, stream>>>(buf0, leading);
    assoc_kernel<<<1, 64, 0, stream>>>(leading, pid, out);
}
